// GRUAutoEncoder_36498632081579
// MI455X (gfx1250) — matrix operand order repaired and checked by reading, not compiled, not run on hardware
//
#include <hip/hip_runtime.h>
#include <hip/hip_bf16.h>

// GRU autoencoder, T=4096, I=512, H=2048, F=256.
// Strategy:
//   - gx/gh/pre computed as large bf16 WMMA GEMMs (weights L2-resident on MI455X's 192MB L2)
//   - xh recurrence (h-independent) as a cheap sequential elementwise scan
//   - h recurrence as a persistent multi-block kernel with a software grid barrier,
//     matvecs over pre-transposed bf16 U from L2 (25MB/step, the true serial bottleneck)
//   - decoder outputs batched into one WMMA GEMM writing d_out directly
// GEMM: each wave computes a 32x64 output tile (2 M-tiles x 4 N-tiles, 8 wmma per k-step)
// with A rows clamped (not masked) so no exec-mask churn in the inner loop.

#define T_LEN  4096
#define I_DIM  512
#define H_DIM  2048
#define F_DIM  256
#define H3_DIM 6144

typedef __attribute__((ext_vector_type(16))) __bf16 v16bf;
typedef __attribute__((ext_vector_type(8)))  float  v8f;
typedef __attribute__((ext_vector_type(4)))  unsigned int u32x4;

union FragU { v16bf v; u32x4 q[2]; };

__device__ __forceinline__ unsigned short f2bf(float f) {
    unsigned u = __float_as_uint(f);
    u += 0x7fffu + ((u >> 16) & 1u);           // round-to-nearest-even
    return (unsigned short)(u >> 16);
}
__device__ __forceinline__ float bf2f(unsigned s) { return __uint_as_float(s << 16); }
__device__ __forceinline__ float sigmoidf(float x) { return 1.f / (1.f + expf(-x)); }

// ---------------------------------------------------------------- mean over T
__global__ void mean_kernel(const float* __restrict__ seq, const float* __restrict__ mask,
                            float* __restrict__ mean) {
    int i = blockIdx.x * blockDim.x + threadIdx.x;
    if (i >= I_DIM) return;
    float s = 0.f, c = 0.f;
    for (int t = 0; t < T_LEN; ++t) {
        s += seq[(long)t * I_DIM + i];
        c += (mask[(long)t * I_DIM + i] != 0.f) ? 1.f : 0.f;
    }
    mean[i] = s / (1e-6f + c);
}

// -------------------------------------------- fp32 -> bf16 row-major (with placement)
__global__ void cast_rows(const float* __restrict__ src, int rows, int cols,
                          unsigned short* __restrict__ dst, int dstLd, int dstColOff) {
    long idx = (long)blockIdx.x * blockDim.x + threadIdx.x;
    long total = (long)rows * cols;
    if (idx >= total) return;
    int c = (int)(idx % cols), r = (int)(idx / cols);
    dst[(long)r * dstLd + dstColOff + c] = f2bf(src[(long)r * cols + c]);
}

// -------------------------------------------- fp32 (K x N) -> bf16 transposed (N x K)
__global__ void transpose_cast(const float* __restrict__ src, int srcRows, int srcLd,
                               int colBegin, int nCols,
                               unsigned short* __restrict__ dst, int dstLd, int dstRowOff) {
    long idx = (long)blockIdx.x * blockDim.x + threadIdx.x;
    long total = (long)srcRows * nCols;
    if (idx >= total) return;
    int c = (int)(idx % nCols), r = (int)(idx / nCols);
    dst[(long)c * dstLd + dstRowOff + r] = f2bf(src[(long)r * srcLd + colBegin + c]);
}

// ---------------------------------------------------------------- WMMA GEMM
// C(MxN,f32) = epi( A(MxK,bf16 row-major) * B (via BT: NxK bf16 row-major) + bias )
// mode 0: +bias              mode 1: exp(-relu(x+bias))
// A-fragment (16x32, 16-bit): lane L -> row (L&15); k = kIter + ((L>>4)<<4) + {0..15}
// B-fragment (32x16, 16-bit): lane L -> col (L&15); k = kIter + ((L>>4)<<4) + {0..15}
// C/D: lane L -> col (L&15); vgpr r -> row ((L>>4)<<3)+r
// Wave tile: 32(M) x 64(N).  Block: 8 waves -> 32 x 512.
__global__ void gemm_bf16(const unsigned short* __restrict__ A, int lda,
                          const unsigned short* __restrict__ BT, int ldb,
                          const float* __restrict__ bias, float* __restrict__ C, int ldc,
                          int M, int N, int K, int mode) {
    const int lane = threadIdx.x & 31;
    const int wave = threadIdx.x >> 5;
    const int mBase = blockIdx.y * 32;
    const int nBase = (blockIdx.x * 8 + wave) * 64;
    if (nBase >= N) return;

    // A rows clamped instead of masked: rows >= M produce garbage only in D rows >= M,
    // which are discarded at the store.  Keeps the inner loop branch-free.
    int aRow0 = mBase + (lane & 15);      if (aRow0 >= M) aRow0 = M - 1;
    int aRow1 = mBase + 16 + (lane & 15); if (aRow1 >= M) aRow1 = M - 1;
    const unsigned short* aPtr0 = A + (long)aRow0 * lda + ((lane >> 4) << 4);
    const unsigned short* aPtr1 = A + (long)aRow1 * lda + ((lane >> 4) << 4);
    const unsigned short* bPtr  = BT + (long)(nBase + (lane & 15)) * ldb + ((lane >> 4) << 4);
    const long bStep = (long)16 * ldb;    // next 16-column tile of B

    v8f acc00 = {0.f,0.f,0.f,0.f,0.f,0.f,0.f,0.f};
    v8f acc01 = acc00, acc02 = acc00, acc03 = acc00;
    v8f acc10 = acc00, acc11 = acc00, acc12 = acc00, acc13 = acc00;

    for (int k = 0; k < K; k += 32) {
        FragU a0, a1, b0, b1, b2, b3;
        a0.q[0] = *(const u32x4*)(aPtr0 + k);
        a0.q[1] = *(const u32x4*)(aPtr0 + k + 8);
        a1.q[0] = *(const u32x4*)(aPtr1 + k);
        a1.q[1] = *(const u32x4*)(aPtr1 + k + 8);
        b0.q[0] = *(const u32x4*)(bPtr + k);
        b0.q[1] = *(const u32x4*)(bPtr + k + 8);
        b1.q[0] = *(const u32x4*)(bPtr + bStep + k);
        b1.q[1] = *(const u32x4*)(bPtr + bStep + k + 8);
        b2.q[0] = *(const u32x4*)(bPtr + 2 * bStep + k);
        b2.q[1] = *(const u32x4*)(bPtr + 2 * bStep + k + 8);
        b3.q[0] = *(const u32x4*)(bPtr + 3 * bStep + k);
        b3.q[1] = *(const u32x4*)(bPtr + 3 * bStep + k + 8);
        acc00 = __builtin_amdgcn_wmma_f32_16x16x32_bf16(false, a0.v, false, b0.v, (short)0, acc00, false, false);
        acc01 = __builtin_amdgcn_wmma_f32_16x16x32_bf16(false, a0.v, false, b1.v, (short)0, acc01, false, false);
        acc02 = __builtin_amdgcn_wmma_f32_16x16x32_bf16(false, a0.v, false, b2.v, (short)0, acc02, false, false);
        acc03 = __builtin_amdgcn_wmma_f32_16x16x32_bf16(false, a0.v, false, b3.v, (short)0, acc03, false, false);
        acc10 = __builtin_amdgcn_wmma_f32_16x16x32_bf16(false, a1.v, false, b0.v, (short)0, acc10, false, false);
        acc11 = __builtin_amdgcn_wmma_f32_16x16x32_bf16(false, a1.v, false, b1.v, (short)0, acc11, false, false);
        acc12 = __builtin_amdgcn_wmma_f32_16x16x32_bf16(false, a1.v, false, b2.v, (short)0, acc12, false, false);
        acc13 = __builtin_amdgcn_wmma_f32_16x16x32_bf16(false, a1.v, false, b3.v, (short)0, acc13, false, false);
    }

    v8f accs[8] = {acc00, acc01, acc02, acc03, acc10, acc11, acc12, acc13};
#pragma unroll
    for (int mt = 0; mt < 2; ++mt) {
#pragma unroll
        for (int nt = 0; nt < 4; ++nt) {
            const v8f acc = accs[mt * 4 + nt];
            const int col = nBase + nt * 16 + (lane & 15);
            const float bv = bias ? bias[col] : 0.f;
#pragma unroll
            for (int r = 0; r < 8; ++r) {
                int row = mBase + mt * 16 + ((lane >> 4) << 3) + r;
                if (row < M) {
                    float v = acc[r] + bv;
                    if (mode == 1) v = expf(-fmaxf(v, 0.f));
                    C[(long)row * ldc + col] = v;
                }
            }
        }
    }
}

// ---------------------------------------------------------------- xh chain (h-independent)
__global__ void xh_scan(const float* __restrict__ seq, const float* __restrict__ mask,
                        const float* __restrict__ GX, const float* __restrict__ prex0,
                        const float* __restrict__ mean,
                        unsigned short* __restrict__ AcatE, unsigned short* __restrict__ AcatD) {
    int i = threadIdx.x;                 // 512 threads, one per feature
    float prex = prex0[i];
    float mn = mean[i];
    for (int t = 0; t < T_LEN; ++t) {    // encoder order
        long o = (long)t * I_DIM + i;
        float m = mask[o], x = seq[o], g = GX[o];
        float xh = m * x + (1.f - m) * (g * prex + (1.f - g) * mn);
        AcatE[(long)t * 1024 + i] = f2bf(xh);
        prex = xh;
    }
    for (int s = 0; s < T_LEN - 1; ++s) { // decoder order: t = 4094-s, stored at row t
        int t = T_LEN - 2 - s;
        long o = (long)t * I_DIM + i;
        float m = mask[o], x = seq[o], g = GX[o];
        float xh = m * x + (1.f - m) * (g * prex + (1.f - g) * mn);
        AcatD[(long)t * 1024 + i] = f2bf(xh);
        prex = xh;
    }
}

// ---------------------------------------------------------------- scan state init
__global__ void init_scan_state(const float* __restrict__ h0, float* __restrict__ hcur,
                                unsigned* __restrict__ barCnt) {
    int i = blockIdx.x * blockDim.x + threadIdx.x;
    if (i < H_DIM) hcur[i] = h0[i];
    if (i == 0) *barCnt = 0u;
}
__global__ void reset_bar(unsigned* __restrict__ barCnt) {
    if (threadIdx.x == 0 && blockIdx.x == 0) *barCnt = 0u;
}

// ---------------------------------------------------------------- persistent h-recurrence
__device__ __forceinline__ void grid_sync(unsigned* cnt, unsigned target) {
    __syncthreads();
    if (threadIdx.x == 0) {
        __threadfence();
        __hip_atomic_fetch_add(cnt, 1u, __ATOMIC_ACQ_REL, __HIP_MEMORY_SCOPE_AGENT);
        while (__hip_atomic_load(cnt, __ATOMIC_ACQUIRE, __HIP_MEMORY_SCOPE_AGENT) < target)
            __builtin_amdgcn_s_sleep(2);
        __threadfence();
    }
    __syncthreads();
}

__global__ void rec_scan(const float* __restrict__ PRE, const float* __restrict__ GH,
                         const float* __restrict__ dts,
                         const unsigned short* __restrict__ U1T,   // 4096 x 2048 bf16
                         const unsigned short* __restrict__ U2T,   // 2048 x 2048 bf16
                         float* __restrict__ hcur, float* __restrict__ zbuf,
                         float* __restrict__ rhd,
                         unsigned short* __restrict__ HPOSTbf,     // decoder only
                         float* __restrict__ hlast,                // encoder only
                         int nSteps, int encoder, unsigned* __restrict__ barCnt) {
    __shared__ float hd_lds[H_DIM];
    __shared__ float rhd_lds[H_DIM];
    const int tid = threadIdx.x;
    const int lane = tid & 31;
    const int wavesPerBlock = blockDim.x >> 5;
    const int gwave = blockIdx.x * wavesPerBlock + (tid >> 5);
    const int totalWaves = gridDim.x * wavesPerBlock;
    const unsigned nb = gridDim.x;
    unsigned epoch = 0;

    for (int step = 0; step < nSteps; ++step) {
        const int t = encoder ? step : (nSteps - 1 - step);
        const float* pre = PRE + (long)t * H3_DIM;
        const float* gh  = GH  + (long)t * H_DIM;

        // ---- phase A: hd = gh*h; z = sig(pre0 + hd@U1[:, :2048]); r*hd via cols 2048..4095
        for (int k = tid; k < H_DIM; k += blockDim.x) hd_lds[k] = gh[k] * hcur[k];
        __syncthreads();
        for (int col = gwave; col < 2 * H_DIM; col += totalWaves) {
            const unsigned* up = (const unsigned*)(U1T + (long)col * H_DIM);
            if (col + totalWaves < 2 * H_DIM)
                __builtin_prefetch(U1T + (long)(col + totalWaves) * H_DIM + lane * 32, 0, 1);
            float acc = 0.f;
            for (int i = lane; i < H_DIM / 2; i += 32) {
                unsigned pk = up[i];
                acc += bf2f(pk & 0xffffu) * hd_lds[2 * i] + bf2f(pk >> 16) * hd_lds[2 * i + 1];
            }
            for (int off = 16; off; off >>= 1) acc += __shfl_xor(acc, off, 32);
            if (lane == 0) {
                float v = sigmoidf(pre[col] + acc);
                if (col < H_DIM) zbuf[col] = v;                       // z
                else             rhd[col - H_DIM] = v * hd_lds[col - H_DIM]; // r*hd
            }
        }
        grid_sync(barCnt, (++epoch) * nb);

        // ---- phase B: htil = tanh(pre2 + (r*hd)@U2); h += dt*dh; emit h_post
        for (int k = tid; k < H_DIM; k += blockDim.x) rhd_lds[k] = rhd[k];
        __syncthreads();
        const float dtv = dts[t];
        for (int col = gwave; col < H_DIM; col += totalWaves) {
            const unsigned* up = (const unsigned*)(U2T + (long)col * H_DIM);
            if (col + totalWaves < H_DIM)
                __builtin_prefetch(U2T + (long)(col + totalWaves) * H_DIM + lane * 32, 0, 1);
            float acc = 0.f;
            for (int i = lane; i < H_DIM / 2; i += 32) {
                unsigned pk = up[i];
                acc += bf2f(pk & 0xffffu) * rhd_lds[2 * i] + bf2f(pk >> 16) * rhd_lds[2 * i + 1];
            }
            for (int off = 16; off; off >>= 1) acc += __shfl_xor(acc, off, 32);
            if (lane == 0) {
                float htil = tanhf(pre[2 * H_DIM + col] + acc);
                float hdj = hd_lds[col];
                float zj = zbuf[col];
                float dh = (1.f - zj) * (htil - hdj);
                float hpost = zj * hdj + (1.f - zj) * htil;
                hcur[col] += dtv * dh;
                if (HPOSTbf) HPOSTbf[(long)t * H_DIM + col] = f2bf(hpost);
                else if (hlast && t == nSteps - 1) hlast[col] = hpost;
            }
        }
        grid_sync(barCnt, (++epoch) * nb);
    }
}

// -------------------------------------- encoder->decoder bridge (relu, We, Wd, tiny)
__global__ void feature_kernel(const float* __restrict__ hlast,
                               const float* __restrict__ We, const float* __restrict__ be,
                               const float* __restrict__ Wd, const float* __restrict__ bd,
                               float* __restrict__ hcur) {
    __shared__ float feat[F_DIM];
    int tid = threadIdx.x;              // 256 threads
    float a = 0.f;
    for (int k = 0; k < H_DIM; ++k)
        a += fmaxf(hlast[k], 0.f) * We[(long)k * F_DIM + tid];
    feat[tid] = fmaxf(a + be[tid], 0.f);
    __syncthreads();
    for (int j = tid; j < H_DIM; j += blockDim.x) {
        float s = 0.f;
        for (int f = 0; f < F_DIM; ++f) s += feat[f] * Wd[(long)f * H_DIM + j];
        hcur[j] = s + bd[j];
    }
}

// ================================================================ launch
extern "C" void kernel_launch(void* const* d_in, const int* in_sizes, int n_in,
                              void* d_out, int out_size, void* d_ws, size_t ws_size,
                              hipStream_t stream) {
    (void)in_sizes; (void)n_in; (void)out_size; (void)ws_size;
    const float* seq   = (const float*)d_in[0];
    const float* mask  = (const float*)d_in[1];
    const float* delta = (const float*)d_in[2];
    const float* dt    = (const float*)d_in[3];
    const float* h0    = (const float*)d_in[4];
    /* d_in[5] = dh0 : dead in the recurrence */
    const float* prex0 = (const float*)d_in[6];
    const float* Wgx   = (const float*)d_in[7];
    const float* bgx   = (const float*)d_in[8];
    const float* Wgh   = (const float*)d_in[9];
    const float* bgh   = (const float*)d_in[10];
    const float* W     = (const float*)d_in[11];
    const float* U     = (const float*)d_in[12];
    const float* V     = (const float*)d_in[13];
    const float* b     = (const float*)d_in[14];
    const float* We    = (const float*)d_in[15];
    const float* be    = (const float*)d_in[16];
    const float* Wd    = (const float*)d_in[17];
    const float* bd    = (const float*)d_in[18];
    const float* Wo    = (const float*)d_in[19];
    const float* bo    = (const float*)d_in[20];
    float* out = (float*)d_out;

    char* ws = (char*)d_ws;
    size_t off = 0;
    auto alloc = [&](size_t bytes) -> char* {
        char* p = ws + off;
        off = (off + bytes + 255) & ~(size_t)255;
        return p;
    };

    float* GX   = (float*)alloc((size_t)T_LEN * I_DIM * 4);           // 8 MB
    float* GH   = (float*)alloc((size_t)T_LEN * H_DIM * 4);           // 32 MB
    float* PRE  = (float*)alloc((size_t)T_LEN * H3_DIM * 4);          // 96 MB (enc then dec)
    unsigned short* Dbf   = (unsigned short*)alloc((size_t)T_LEN * I_DIM * 2);
    unsigned short* AcatE = (unsigned short*)alloc((size_t)T_LEN * 1024 * 2);
    unsigned short* AcatD = (unsigned short*)alloc((size_t)(T_LEN - 1) * 1024 * 2);
    unsigned short* WgxT  = (unsigned short*)alloc((size_t)I_DIM * I_DIM * 2);
    unsigned short* WghT  = (unsigned short*)alloc((size_t)H_DIM * I_DIM * 2);
    unsigned short* WVT   = (unsigned short*)alloc((size_t)H3_DIM * 1024 * 2);
    unsigned short* U1T   = (unsigned short*)alloc((size_t)4096 * H_DIM * 2);  // 16 MB
    unsigned short* U2T   = (unsigned short*)alloc((size_t)H_DIM * H_DIM * 2); // 8 MB
    unsigned short* WoT   = (unsigned short*)alloc((size_t)I_DIM * H_DIM * 2);
    unsigned short* HPOST = (unsigned short*)alloc((size_t)(T_LEN - 1) * H_DIM * 2); // 16 MB
    float* mean  = (float*)alloc(I_DIM * 4);
    float* hcur  = (float*)alloc(H_DIM * 4);
    float* zbuf  = (float*)alloc(H_DIM * 4);
    float* rhdb  = (float*)alloc(H_DIM * 4);
    float* hlast = (float*)alloc(H_DIM * 4);
    unsigned* barCnt = (unsigned*)alloc(256);

    auto grid1 = [](long n) { return (unsigned)((n + 255) / 256); };

    // 0) mean over time
    mean_kernel<<<grid1(I_DIM), 256, 0, stream>>>(seq, mask, mean);

    // 1) bf16 conversions / weight transposes (one-time)
    cast_rows<<<grid1((long)T_LEN * I_DIM), 256, 0, stream>>>(delta, T_LEN, I_DIM, Dbf, I_DIM, 0);
    cast_rows<<<grid1((long)T_LEN * I_DIM), 256, 0, stream>>>(mask, T_LEN, I_DIM, AcatE, 1024, I_DIM);
    cast_rows<<<grid1((long)(T_LEN - 1) * I_DIM), 256, 0, stream>>>(mask, T_LEN - 1, I_DIM, AcatD, 1024, I_DIM);
    transpose_cast<<<grid1((long)I_DIM * I_DIM), 256, 0, stream>>>(Wgx, I_DIM, I_DIM, 0, I_DIM, WgxT, I_DIM, 0);
    transpose_cast<<<grid1((long)I_DIM * H_DIM), 256, 0, stream>>>(Wgh, I_DIM, H_DIM, 0, H_DIM, WghT, I_DIM, 0);
    transpose_cast<<<grid1((long)I_DIM * H3_DIM), 256, 0, stream>>>(W, I_DIM, H3_DIM, 0, H3_DIM, WVT, 1024, 0);
    transpose_cast<<<grid1((long)I_DIM * H3_DIM), 256, 0, stream>>>(V, I_DIM, H3_DIM, 0, H3_DIM, WVT, 1024, I_DIM);
    transpose_cast<<<grid1((long)H_DIM * 4096), 256, 0, stream>>>(U, H_DIM, H3_DIM, 0, 4096, U1T, H_DIM, 0);
    transpose_cast<<<grid1((long)H_DIM * H_DIM), 256, 0, stream>>>(U, H_DIM, H3_DIM, 4096, H_DIM, U2T, H_DIM, 0);
    transpose_cast<<<grid1((long)H_DIM * I_DIM), 256, 0, stream>>>(Wo, H_DIM, I_DIM, 0, I_DIM, WoT, H_DIM, 0);

    // 2) GX = exp(-relu(delta@Wgx+bgx)), GH = exp(-relu(delta@Wgh+bgh))  [WMMA]
    gemm_bf16<<<dim3(1, T_LEN / 32), 256, 0, stream>>>(
        Dbf, I_DIM, WgxT, I_DIM, bgx, GX, I_DIM, T_LEN, I_DIM, I_DIM, 1);
    gemm_bf16<<<dim3(H_DIM / 512, T_LEN / 32), 256, 0, stream>>>(
        Dbf, I_DIM, WghT, I_DIM, bgh, GH, H_DIM, T_LEN, H_DIM, I_DIM, 1);

    // 3) xh chains (encoder then decoder), bf16 into [XH|M] activation matrices
    xh_scan<<<1, I_DIM, 0, stream>>>(seq, mask, GX, prex0, mean, AcatE, AcatD);

    // 4) PRE_enc = [XH|M] @ [W;V] + b   [WMMA, K=1024]
    gemm_bf16<<<dim3(H3_DIM / 512, T_LEN / 32), 256, 0, stream>>>(
        AcatE, 1024, WVT, 1024, b, PRE, H3_DIM, T_LEN, H3_DIM, 1024, 0);

    // 5) encoder h-recurrence (persistent, grid-barriered)
    init_scan_state<<<grid1(H_DIM), 256, 0, stream>>>(h0, hcur, barCnt);
    rec_scan<<<64, 256, 0, stream>>>(PRE, GH, dt, U1T, U2T, hcur, zbuf, rhdb,
                                     nullptr, hlast, T_LEN, 1, barCnt);

    // 6) bridge: feature = relu(relu(h_post)@We+be); h_dec0 = feature@Wd+bd
    feature_kernel<<<1, F_DIM, 0, stream>>>(hlast, We, be, Wd, bd, hcur);

    // 7) PRE_dec (reuse PRE buffer, rows indexed by t)   [WMMA]
    gemm_bf16<<<dim3(H3_DIM / 512, (T_LEN - 1 + 31) / 32), 256, 0, stream>>>(
        AcatD, 1024, WVT, 1024, b, PRE, H3_DIM, T_LEN - 1, H3_DIM, 1024, 0);

    // 8) decoder h-recurrence (t = 4094..0), h_post rows stored at row t
    reset_bar<<<1, 64, 0, stream>>>(barCnt);
    rec_scan<<<64, 256, 0, stream>>>(PRE, GH, dt, U1T, U2T, hcur, zbuf, rhdb,
                                     HPOST, nullptr, T_LEN - 1, 0, barCnt);

    // 9) out = HPOST @ Wo + bo  -> already in output row order   [WMMA]
    gemm_bf16<<<dim3(1, (T_LEN - 1 + 31) / 32), 256, 0, stream>>>(
        HPOST, H_DIM, WoT, H_DIM, bo, out, I_DIM, T_LEN - 1, I_DIM, H_DIM, 0);
}
